// MultiZoomFieldAttention_12008728560002
// MI455X (gfx1250) — matrix operand order repaired and checked by reading, not compiled, not run on hardware
//
#include <hip/hip_runtime.h>

// ---------------------------------------------------------------------------
// MI455X (gfx1250) implementation.
//  - wave32, v_wmma_f32_16x16x32_bf16 everywhere (K=32 == HD).
//  - Flash attention: 768 q x 6912 k per head, streamed in 128-key blocks
//    (16 WMMA per wave per block; 3 barriers per 128 keys).
//  - K/V projected once per node (768) and gathered via adjc (9x reuse).
//  - K block staged via GLOBAL_LOAD_ASYNC_TO_LDS (ASYNCcnt) when the
//    toolchain exposes the builtin; sync fallback otherwise.
//  - bf16 storage for Q/K/V and weights; f32 accumulation + softmax/LN/GELU.
// ---------------------------------------------------------------------------

typedef __bf16 v16bf __attribute__((ext_vector_type(16)));
typedef float  v8f   __attribute__((ext_vector_type(8)));
typedef unsigned int v4u __attribute__((ext_vector_type(4)));
typedef int    v4i   __attribute__((ext_vector_type(4)));
typedef unsigned short u16;
typedef unsigned int   u32;

#define NNODE 768
#define NHADJ 9
#define NKEY  (NNODE * NHADJ)   // 6912
#define NROW  12288             // n*s feature rows
#define KB    128               // keys per attention block (6912 = 54*128)
#define QSCALE 0.17677669529663687f  // 1/sqrt(32), folded into Q

#if defined(__has_builtin)
# if __has_builtin(__builtin_amdgcn_global_load_async_to_lds_b128) && \
     __has_builtin(__builtin_amdgcn_s_wait_asynccnt)
#  define USE_ASYNC 1
# endif
#endif
#ifndef USE_ASYNC
# define USE_ASYNC 0
#endif
#define AS1 __attribute__((address_space(1)))
#define AS3 __attribute__((address_space(3)))

union V16 { v16bf v; v4u q[2]; };

__device__ inline u16 f2bf(float f) {
  union { float f; u32 u; } c; c.f = f;
  u32 u = c.u;
  return (u16)((u + 0x7FFFu + ((u >> 16) & 1u)) >> 16);
}

// Build a v16bf operand from two 16-byte chunks (callers pass hi = lo + 8:
// contiguous K = 0..15 or 16..31 per lane half).
__device__ inline v16bf load_ab(const u16* lo, const u16* hi) {
  V16 t;
  t.q[0] = *(const v4u*)lo;
  t.q[1] = *(const v4u*)hi;
  return t.v;
}

// ---------------------------------------------------------------------------
// Kernel 1: per-node projections.
// xs = (A1 (x) A2)^T * xr  (16x16 spatial, shared by q/k/v), then
// Q/K/V = xs @ W{q,k,v} via WMMA. Outputs per-head layout [head][node][32].
// 192 blocks x 128 threads, 1 node per wave.
// ---------------------------------------------------------------------------
__global__ __launch_bounds__(128) void proj_kernel(
    const float* __restrict__ x, const float* __restrict__ A1,
    const float* __restrict__ A2, const float* __restrict__ Wq,
    const float* __restrict__ Wk, const float* __restrict__ Wv,
    u16* __restrict__ Qp, u16* __restrict__ Kp, u16* __restrict__ Vp)
{
  __shared__ float S2[16][16];          // spatial transform (A1 (x) A2)
  __shared__ u16   Wt[3][64][72];       // W^T, bf16: Wt[p][out][in]
  __shared__ float xrb[4][16][68];      // per-wave input tile
  __shared__ u16   xsb[4][16][40];      // per-wave spatial-transformed tile

  const int tid  = threadIdx.x;
  const int w    = tid >> 5;
  const int lane = tid & 31;

  // S2[s][r] = A1[x][X] * A2[y][Y], s=(x,y) r=(X,Y)
  for (int idx = tid; idx < 256; idx += 128) {
    int s = idx >> 4, r = idx & 15;
    S2[s][r] = A1[(s >> 2) * 4 + (r >> 2)] * A2[(s & 3) * 4 + (r & 3)];
  }
  // Transposed bf16 weights: W is [in(64)][out(64)] row-major
  for (int idx = tid; idx < 4096; idx += 128) {
    int c = idx >> 6, o = idx & 63;
    Wt[0][o][c] = f2bf(Wq[idx]);
    Wt[1][o][c] = f2bf(Wk[idx]);
    Wt[2][o][c] = f2bf(Wv[idx]);
  }

  const int node = blockIdx.x * 4 + w;
  const float* src = x + node * 1024;   // 16 x 64 tile
#pragma unroll
  for (int i = 0; i < 32; ++i) {
    int e = lane * 32 + i;              // 32 contiguous floats per lane
    xrb[w][e >> 6][e & 63] = src[e];
  }
  __syncthreads();

  // xs[r][c] = sum_s S2[s][r] * xr[s][c]; each lane owns 2 columns
#pragma unroll
  for (int half = 0; half < 2; ++half) {
    int c = lane + half * 32;
    float col[16];
#pragma unroll
    for (int s = 0; s < 16; ++s) col[s] = xrb[w][s][c];
#pragma unroll
    for (int r = 0; r < 16; ++r) {
      float acc = 0.f;
#pragma unroll
      for (int s = 0; s < 16; ++s) acc += S2[s][r] * col[s];
      xsb[w][r][c] = f2bf(acc);
    }
  }
  __syncthreads();

  // A-matrix (16x64, two K=32 chunks) per WMMA layout
  const int arow = lane & 15;
  const int alo  = (lane < 16) ? 0 : 16;
  v16bf a0 = load_ab(&xsb[w][arow][alo],      &xsb[w][arow][alo + 8]);
  v16bf a1 = load_ab(&xsb[w][arow][alo + 32], &xsb[w][arow][alo + 40]);

  u16* outs[3] = { Qp, Kp, Vp };
  const int bcol = lane & 15;
  const int bk   = (lane < 16) ? 0 : 16;
#pragma unroll
  for (int p = 0; p < 3; ++p) {
#pragma unroll
    for (int nt = 0; nt < 4; ++nt) {
      const int o = nt * 16 + bcol;
      v16bf b0 = load_ab(&Wt[p][o][bk],      &Wt[p][o][bk + 8]);
      v16bf b1 = load_ab(&Wt[p][o][32 + bk], &Wt[p][o][32 + bk + 8]);
      v8f acc = {};
      acc = __builtin_amdgcn_wmma_f32_16x16x32_bf16(false, a0, false, b0,
                                                    (short)0, acc, false, false);
      acc = __builtin_amdgcn_wmma_f32_16x16x32_bf16(false, a1, false, b1,
                                                    (short)0, acc, false, false);
#pragma unroll
      for (int r = 0; r < 8; ++r) {
        int M = r + ((lane < 16) ? 0 : 8);   // spatial row 0..15
        int c = nt * 16 + bcol;              // channel 0..63
        int head = M * 2 + (c >> 5);         // flat (r*64+c) -> (head, d)
        int d = c & 31;
        float v = acc[r];
        if (p == 0) v *= QSCALE;             // fold 1/sqrt(HD) into Q only
        outs[p][(head * NNODE + node) * 32 + d] = f2bf(v);
      }
    }
  }
}

// ---------------------------------------------------------------------------
// Kernel 2: flash attention. Grid (32 heads, 12 q-blocks) x 128 threads.
// Each wave owns a 16-query tile; workgroup shares gather-staged K/V blocks.
// Per 128-key block: 8 WMMA (QK^T) + online softmax + 8 WMMA (P@V).
// ---------------------------------------------------------------------------
__global__ __launch_bounds__(128) void attn_kernel(
    const u16* __restrict__ Qp, const u16* __restrict__ Kp,
    const u16* __restrict__ Vp, const int* __restrict__ adjc,
    float* __restrict__ Ob)
{
  __shared__ u16 Kb[KB][40];        // K block: [key][d]
  __shared__ u16 Vt[32][136];       // V block transposed: [d][key]
  __shared__ u16 Pb[4][16][136];    // per-wave P round-trip (C -> A layout)

  const int h    = blockIdx.x;
  const int qb   = blockIdx.y;
  const int tid  = threadIdx.x;
  const int w    = tid >> 5;
  const int lane = tid & 31;

  const int arow = lane & 15;
  const int alo  = (lane < 16) ? 0 : 16;
  const int bcol = lane & 15;
  const int bk   = (lane < 16) ? 0 : 16;

  // Load this wave's Q tile (16 x 32) straight into A layout
  const int  qrow = qb * 64 + w * 16 + arow;
  const u16* qptr = Qp + (h * NNODE + qrow) * 32;
  v16bf qa = load_ab(qptr + alo, qptr + alo + 8);

  v8f acc0 = {}, acc1 = {};
  float mrow[8], lrow[8];
#pragma unroll
  for (int r = 0; r < 8; ++r) { mrow[r] = -1e30f; lrow[r] = 0.f; }

  for (int kb = 0; kb < NKEY / KB; ++kb) {
    __syncthreads();                          // previous block fully consumed
    // ---- stage: one key row (32 bf16) per thread, gathered via adjc ----
    const int  node = adjc[kb * KB + tid];
    const u16* ks = Kp + (h * NNODE + node) * 32;
#if USE_ASYNC
    {
      AS1 v4i* gp = (AS1 v4i*)(void*)ks;
      AS3 v4i* lp = (AS3 v4i*)(void*)&Kb[tid][0];
      __builtin_amdgcn_global_load_async_to_lds_b128(gp, lp, 0, 0);
      __builtin_amdgcn_global_load_async_to_lds_b128(gp, lp, 16, 0);
      __builtin_amdgcn_global_load_async_to_lds_b128(gp, lp, 32, 0);
      __builtin_amdgcn_global_load_async_to_lds_b128(gp, lp, 48, 0);
    }
#else
    {
      v4u* kdst = (v4u*)&Kb[tid][0];
#pragma unroll
      for (int c4 = 0; c4 < 4; ++c4) kdst[c4] = ((const v4u*)ks)[c4];
    }
#endif
    const u16* vsrc = Vp + (h * NNODE + node) * 32;
#pragma unroll
    for (int c4 = 0; c4 < 4; ++c4) {          // V transposed into LDS
      v4u vv = ((const v4u*)vsrc)[c4];
      const u16* ve = (const u16*)&vv;
#pragma unroll
      for (int e = 0; e < 8; ++e) Vt[c4 * 8 + e][tid] = ve[e];
    }
#if USE_ASYNC
    __builtin_amdgcn_s_wait_asynccnt(0);
#endif
    __syncthreads();

    // ---- S = Q @ K^T over 8 key tiles ----
    v8f s[8];
#pragma unroll
    for (int t = 0; t < 8; ++t) {
      v16bf kv = load_ab(&Kb[t * 16 + bcol][bk], &Kb[t * 16 + bcol][bk + 8]);
      v8f z = {};
      s[t] = __builtin_amdgcn_wmma_f32_16x16x32_bf16(false, qa, false, kv,
                                                     (short)0, z, false, false);
    }

    // ---- online softmax across 128 keys per row ----
#pragma unroll
    for (int r = 0; r < 8; ++r) {
      float mx = s[0][r];
#pragma unroll
      for (int t = 1; t < 8; ++t) mx = fmaxf(mx, s[t][r]);
#pragma unroll
      for (int m = 1; m < 16; m <<= 1) mx = fmaxf(mx, __shfl_xor(mx, m, 32));
      float newm = fmaxf(mrow[r], mx);
      float ps = 0.f;
      int M = r + ((lane < 16) ? 0 : 8);
#pragma unroll
      for (int t = 0; t < 8; ++t) {
        float p = __expf(s[t][r] - newm);
        ps += p;
        Pb[w][M][t * 16 + bcol] = f2bf(p);
      }
#pragma unroll
      for (int m = 1; m < 16; m <<= 1) ps += __shfl_xor(ps, m, 32);
      float corr = __expf(mrow[r] - newm);
      lrow[r] = lrow[r] * corr + ps;
      mrow[r] = newm;
      acc0[r] *= corr;
      acc1[r] *= corr;
    }
    __syncthreads();                          // publish P (and fence TBAA)

    // ---- O += P @ V over 4 K-chunks of 32 keys ----
#pragma unroll
    for (int c = 0; c < 4; ++c) {
      v16bf pa = load_ab(&Pb[w][arow][c * 32 + alo],
                         &Pb[w][arow][c * 32 + alo + 8]);
      v16bf v0 = load_ab(&Vt[bcol][c * 32 + bk],
                         &Vt[bcol][c * 32 + bk + 8]);
      v16bf v1 = load_ab(&Vt[16 + bcol][c * 32 + bk],
                         &Vt[16 + bcol][c * 32 + bk + 8]);
      acc0 = __builtin_amdgcn_wmma_f32_16x16x32_bf16(false, pa, false, v0,
                                                     (short)0, acc0, false, false);
      acc1 = __builtin_amdgcn_wmma_f32_16x16x32_bf16(false, pa, false, v1,
                                                     (short)0, acc1, false, false);
    }
  }

  // Normalize and scatter: o[q][head][d] -> row nf = q*16 + h/2, col (h&1)*32+d
#pragma unroll
  for (int r = 0; r < 8; ++r) {
    int M = r + ((lane < 16) ? 0 : 8);
    int q = qb * 64 + w * 16 + M;
    float inv = 1.0f / lrow[r];
    int nf = q * 16 + (h >> 1);
    int cb = (h & 1) * 32;
    Ob[nf * 64 + cb + bcol]      = acc0[r] * inv;
    Ob[nf * 64 + cb + 16 + bcol] = acc1[r] * inv;
  }
}

// ---------------------------------------------------------------------------
// Kernel 3: fused residual + LayerNorm + MLP (64->128 GELU ->64) + residual.
// 192 blocks x 128 threads; 64 rows per block, 16-row WMMA tiles per wave.
// ---------------------------------------------------------------------------
__global__ __launch_bounds__(128) void mlp_kernel(
    const float* __restrict__ x, const float* __restrict__ Ob,
    const float* __restrict__ gamma, const float* __restrict__ ln_g,
    const float* __restrict__ ln_b, const float* __restrict__ W1,
    const float* __restrict__ b1, const float* __restrict__ W2,
    const float* __restrict__ b2, const float* __restrict__ gamma_mlp,
    float* __restrict__ out)
{
  __shared__ u16 W1t[128][72];      // W1^T bf16: [out128][in64]
  __shared__ u16 W2t[64][136];      // W2^T bf16: [out64][in128]
  __shared__ u16 hbuf[64][72];      // LN output, bf16
  __shared__ u16 gbuf[4][16][136];  // per-wave GELU output round-trip

  const int tid  = threadIdx.x;
  const int w    = tid >> 5;
  const int lane = tid & 31;
  const int rbase = blockIdx.x * 64;

  for (int idx = tid; idx < 8192; idx += 128) {   // W1 is [64][128]
    W1t[idx & 127][idx >> 7] = f2bf(W1[idx]);
  }
  for (int idx = tid; idx < 8192; idx += 128) {   // W2 is [128][64]
    W2t[idx & 63][idx >> 6] = f2bf(W2[idx]);
  }

  // Residual + LayerNorm: 2 threads per row (adjacent lanes)
  {
    int row  = tid >> 1;
    int half = tid & 1;
    int grow = rbase + row;
    float vals[32];
    float s = 0.f, ss = 0.f;
#pragma unroll
    for (int i = 0; i < 32; ++i) {
      int c = half * 32 + i;
      float v = x[grow * 64 + c] + gamma[c] * Ob[grow * 64 + c];
      vals[i] = v;
      s += v; ss += v * v;
    }
    s  += __shfl_xor(s, 1, 32);
    ss += __shfl_xor(ss, 1, 32);
    float mu  = s * (1.f / 64.f);
    float var = ss * (1.f / 64.f) - mu * mu;
    float rs  = rsqrtf(var + 1e-5f);
#pragma unroll
    for (int i = 0; i < 32; ++i) {
      int c = half * 32 + i;
      hbuf[row][c] = f2bf((vals[i] - mu) * rs * ln_g[c] + ln_b[c]);
    }
  }
  __syncthreads();

  const int arow = lane & 15;
  const int alo  = (lane < 16) ? 0 : 16;
  const int bcol = lane & 15;
  const int bk   = (lane < 16) ? 0 : 16;
  const int lr   = w * 16 + arow;

  // GEMM1: (16x64) @ (64x128) + b1, GELU
  v16bf a0 = load_ab(&hbuf[lr][alo],      &hbuf[lr][alo + 8]);
  v16bf a1 = load_ab(&hbuf[lr][alo + 32], &hbuf[lr][alo + 40]);
#pragma unroll
  for (int nt = 0; nt < 8; ++nt) {
    const int o = nt * 16 + bcol;
    v16bf b0 = load_ab(&W1t[o][bk],      &W1t[o][bk + 8]);
    v16bf b1v = load_ab(&W1t[o][32 + bk], &W1t[o][32 + bk + 8]);
    v8f acc = {};
    acc = __builtin_amdgcn_wmma_f32_16x16x32_bf16(false, a0, false, b0,
                                                  (short)0, acc, false, false);
    acc = __builtin_amdgcn_wmma_f32_16x16x32_bf16(false, a1, false, b1v,
                                                  (short)0, acc, false, false);
    const float bias = b1[o];
#pragma unroll
    for (int r = 0; r < 8; ++r) {
      float v = acc[r] + bias;
      float g = 0.5f * v * (1.f + tanhf(0.7978845608f * (v + 0.044715f * v * v * v)));
      int M = r + ((lane < 16) ? 0 : 8);
      gbuf[w][M][o] = f2bf(g);
    }
  }
  __syncthreads();

  // GEMM2: (16x128) @ (128x64) + b2; out = x1 + gamma_mlp * mlp
  v16bf p0 = load_ab(&gbuf[w][arow][alo],      &gbuf[w][arow][alo + 8]);
  v16bf p1 = load_ab(&gbuf[w][arow][32 + alo], &gbuf[w][arow][32 + alo + 8]);
  v16bf p2 = load_ab(&gbuf[w][arow][64 + alo], &gbuf[w][arow][64 + alo + 8]);
  v16bf p3 = load_ab(&gbuf[w][arow][96 + alo], &gbuf[w][arow][96 + alo + 8]);
#pragma unroll
  for (int nt = 0; nt < 4; ++nt) {
    const int o = nt * 16 + bcol;
    v16bf b0 = load_ab(&W2t[o][bk],      &W2t[o][bk + 8]);
    v16bf b1v = load_ab(&W2t[o][32 + bk], &W2t[o][32 + bk + 8]);
    v16bf b2v = load_ab(&W2t[o][64 + bk], &W2t[o][64 + bk + 8]);
    v16bf b3v = load_ab(&W2t[o][96 + bk], &W2t[o][96 + bk + 8]);
    v8f acc = {};
    acc = __builtin_amdgcn_wmma_f32_16x16x32_bf16(false, p0, false, b0,
                                                  (short)0, acc, false, false);
    acc = __builtin_amdgcn_wmma_f32_16x16x32_bf16(false, p1, false, b1v,
                                                  (short)0, acc, false, false);
    acc = __builtin_amdgcn_wmma_f32_16x16x32_bf16(false, p2, false, b2v,
                                                  (short)0, acc, false, false);
    acc = __builtin_amdgcn_wmma_f32_16x16x32_bf16(false, p3, false, b3v,
                                                  (short)0, acc, false, false);
    const float bias = b2[o];
#pragma unroll
    for (int r = 0; r < 8; ++r) {
      int M = r + ((lane < 16) ? 0 : 8);
      int grow = rbase + w * 16 + M;
      float x1 = x[grow * 64 + o] + gamma[o] * Ob[grow * 64 + o];
      out[grow * 64 + o] = x1 + gamma_mlp[o] * (acc[r] + bias);
    }
  }
}

// ---------------------------------------------------------------------------
extern "C" void kernel_launch(void* const* d_in, const int* in_sizes, int n_in,
                              void* d_out, int out_size, void* d_ws, size_t ws_size,
                              hipStream_t stream) {
  const float* x       = (const float*)d_in[0];
  const int*   adjc    = (const int*)d_in[1];
  const float* A1      = (const float*)d_in[2];
  const float* A2      = (const float*)d_in[3];
  const float* Wq      = (const float*)d_in[4];
  const float* Wk      = (const float*)d_in[5];
  const float* Wv      = (const float*)d_in[6];
  const float* gamma   = (const float*)d_in[7];
  const float* ln_g    = (const float*)d_in[8];
  const float* ln_b    = (const float*)d_in[9];
  const float* W1      = (const float*)d_in[10];
  const float* b1      = (const float*)d_in[11];
  const float* W2      = (const float*)d_in[12];
  const float* b2      = (const float*)d_in[13];
  const float* g_mlp   = (const float*)d_in[14];
  float* out = (float*)d_out;

  // Workspace: Qp/Kp/Vp bf16 [32 heads][768 nodes][32], then O f32 [12288][64]
  u16* Qp = (u16*)d_ws;
  u16* Kp = Qp + 32 * NNODE * 32;
  u16* Vp = Kp + 32 * NNODE * 32;
  float* Ob = (float*)(Vp + 32 * NNODE * 32);

  hipLaunchKernelGGL(proj_kernel, dim3(NNODE / 4), dim3(128), 0, stream,
                     x, A1, A2, Wq, Wk, Wv, Qp, Kp, Vp);
  hipLaunchKernelGGL(attn_kernel, dim3(32, NNODE / 64), dim3(128), 0, stream,
                     Qp, Kp, Vp, adjc, Ob);
  hipLaunchKernelGGL(mlp_kernel, dim3(NROW / 64), dim3(128), 0, stream,
                     x, Ob, gamma, ln_g, ln_b, W1, b1, W2, b2, g_mlp, out);
}
